// SelfAttention_20710332301555
// MI455X (gfx1250) — compile-verified, matrix operand order checked by reading, not run on hardware
//
#include <hip/hip_runtime.h>
#include <hip/hip_bf16.h>
#include <math.h>
#include <stdint.h>

typedef __attribute__((ext_vector_type(16))) _Float16 v16h;
typedef __attribute__((ext_vector_type(8)))  _Float16 v8h;
typedef __attribute__((ext_vector_type(8)))  float    v8f;

#define BATCH 8
#define SEQ   2048
#define DIM   1024
#define BS    (BATCH*SEQ)   /* 16384 rows of x */
#define KBLK  32            /* K per v_wmma_f32_16x16x32_f16 */

__device__ __forceinline__ v8f wmma_step(v16h a, v16h b, v8f c) {
  // (neg_a, A, neg_b, B, c_mod, C, reuse_a, reuse_b)
  return __builtin_amdgcn_wmma_f32_16x16x32_f16(false, a, false, b, (short)0, c, false, false);
}

// CDNA5 async global->LDS copy (ASYNCcnt-tracked), portable inline asm.
__device__ __forceinline__ void async_b128(uint32_t lds_off, const void* g) {
  asm volatile("global_load_async_to_lds_b128 %0, %1, off"
               :: "v"(lds_off), "v"((unsigned long long)(uintptr_t)g)
               : "memory");
}
__device__ __forceinline__ void wait_async0() {
  asm volatile("s_wait_asynccnt 0x0" ::: "memory");
}

// ---------------- f32 -> f16 conversion of x (row-major kept) ----------------
__global__ void k_cvt_x(const float* __restrict__ src, _Float16* __restrict__ dst) {
  size_t i = ((size_t)blockIdx.x * blockDim.x + threadIdx.x) * 8;
  const float4* s = reinterpret_cast<const float4*>(src + i);
  float4 a = s[0], b = s[1];
  v8h o;
  o[0] = (_Float16)a.x; o[1] = (_Float16)a.y; o[2] = (_Float16)a.z; o[3] = (_Float16)a.w;
  o[4] = (_Float16)b.x; o[5] = (_Float16)b.y; o[6] = (_Float16)b.z; o[7] = (_Float16)b.w;
  *reinterpret_cast<v8h*>(dst + i) = o;
}

// ------- W (row e, col d) f32 -> packed f16 B-layout [kb][n][32] -------------
__global__ void k_cvt_w_packed(const float* __restrict__ W, _Float16* __restrict__ P) {
  int t  = blockIdx.x * blockDim.x + threadIdx.x;  // DIM*(DIM/32) threads
  int n  = t >> 5;
  int kb = t & 31;
  const float4* s = reinterpret_cast<const float4*>(W + (size_t)n * DIM + kb * 32);
  _Float16* d = P + (size_t)kb * DIM * 32 + (size_t)n * 32;
#pragma unroll
  for (int c = 0; c < 4; ++c) {
    float4 f0 = s[2 * c], f1 = s[2 * c + 1];
    v8h o;
    o[0] = (_Float16)f0.x; o[1] = (_Float16)f0.y; o[2] = (_Float16)f0.z; o[3] = (_Float16)f0.w;
    o[4] = (_Float16)f1.x; o[5] = (_Float16)f1.y; o[6] = (_Float16)f1.z; o[7] = (_Float16)f1.w;
    *reinterpret_cast<v8h*>(d + c * 8) = o;
  }
}

// -------------------- QKV projection: Y = Xf16 * Wt + bias ------------------
// Block 64(M) x 128(N), 8 waves (4 M x 2 N), wave tile 16x64 (4 accums).
// A (64x32, 4KB) and B (128x32, 8KB) tiles double-buffered in LDS via async
// copies. kb-loop NOT unrolled; buffers selected by runtime XOR toggles so the
// accumulators keep one register assignment across the backedge (no movs).
template <bool PACK_V>
__global__ void __launch_bounds__(256)
k_qkv_gemm(const _Float16* __restrict__ xh,
           const _Float16* __restrict__ wP,
           const float*    __restrict__ bias,
           _Float16* __restrict__ outp) {
  __shared__ _Float16 ldsA[2][64 * 32];   // 2 x 4 KB
  __shared__ _Float16 ldsB[2][128 * 32];  // 2 x 8 KB
  const int tid   = threadIdx.x;
  const int lane  = tid & 31;
  const int wv    = tid >> 5;
  const int mw    = wv & 3;
  const int nw    = wv >> 2;
  const int Mb    = blockIdx.x * 64;
  const int Nb    = blockIdx.y * 128;
  const int l15   = lane & 15;
  const int khalf = (lane < 16) ? 0 : 16;
  const uint32_t A_STRIDE = 64 * 32 * 2;    // bytes per A buffer
  const uint32_t B_STRIDE = 128 * 32 * 2;   // bytes per B buffer
  const uint32_t ldsABase = (uint32_t)(uintptr_t)&ldsA[0][0];
  const uint32_t ldsBBase = (uint32_t)(uintptr_t)&ldsB[0][0];
  const _Float16* ldsAp = &ldsA[0][0];
  const _Float16* ldsBp = &ldsB[0][0];

  v8f acc0 = {}, acc1 = {}, acc2 = {}, acc3 = {};

  auto issueAB = [&](int kb, uint32_t wrA, uint32_t wrB) {
    {  // A tile: row = tid/4, 16B chunk = tid%4
      const int row = tid >> 2, ch = tid & 3;
      const _Float16* srcA = xh + (size_t)(Mb + row) * DIM + kb * KBLK + ch * 8;
      async_b128(ldsABase + wrA + (uint32_t)(row * 64 + ch * 16), srcA);
    }
    {  // B tile: col = tid/2, 32B half = tid%2
      const int c = tid >> 1, kk0 = (tid & 1) * 16;
      const _Float16* srcB = wP + (size_t)kb * (DIM * 32) + (size_t)(Nb + c) * 32 + kk0;
      uint32_t offB = ldsBBase + wrB + (uint32_t)(c * 64 + kk0 * 2);
      async_b128(offB, srcB); async_b128(offB + 16, srcB + 8);
    }
  };

  const int NK = DIM / KBLK;
  uint32_t wrA = 0, wrB = 0;          // byte offset of write buffer
  uint32_t rdA = 0, rdB = 0;          // HALF offset of read buffer
  issueAB(0, 0, 0);
  const uint32_t aFragIdx = (uint32_t)((mw * 16 + l15) * 32 + khalf);
  const uint32_t bFragIdx = (uint32_t)(nw * 64 * 32 + khalf);
#pragma clang loop unroll(disable)
  for (int kb = 0; kb < NK; ++kb) {
    wait_async0();
    __syncthreads();
    wrA ^= A_STRIDE; wrB ^= B_STRIDE;
    if (kb + 1 < NK) issueAB(kb + 1, wrA, wrB);
    v16h a = *reinterpret_cast<const v16h*>(ldsAp + rdA + aFragIdx);
    const _Float16* lb = ldsBp + rdB + bFragIdx;
    v16h b0 = *reinterpret_cast<const v16h*>(lb + (0 * 16 + l15) * 32);
    v16h b1 = *reinterpret_cast<const v16h*>(lb + (1 * 16 + l15) * 32);
    v16h b2 = *reinterpret_cast<const v16h*>(lb + (2 * 16 + l15) * 32);
    v16h b3 = *reinterpret_cast<const v16h*>(lb + (3 * 16 + l15) * 32);
    acc0 = wmma_step(a, b0, acc0);
    acc1 = wmma_step(a, b1, acc1);
    acc2 = wmma_step(a, b2, acc2);
    acc3 = wmma_step(a, b3, acc3);
    rdA ^= (A_STRIDE / 2); rdB ^= (B_STRIDE / 2);   // half-element offsets
  }

  const int mrow0 = Mb + mw * 16 + ((lane < 16) ? 0 : 8);
  v8f accs[4] = {acc0, acc1, acc2, acc3};
#pragma unroll
  for (int j = 0; j < 4; ++j) {
    const int n  = Nb + nw * 64 + j * 16 + l15;
    const float bn = bias[n];
    if (PACK_V) {
      const int bidx = mrow0 / SEQ;
      const int s0   = mrow0 % SEQ;
      _Float16* d = outp + (size_t)bidx * SEQ * DIM + (size_t)(s0 >> 5) * DIM * 32 +
                    (size_t)n * 32 + (s0 & 31);
      v8h o;
#pragma unroll
      for (int v = 0; v < 8; ++v) o[v] = (_Float16)(accs[j][v] + bn);
      *reinterpret_cast<v8h*>(d) = o;
    } else {
#pragma unroll
      for (int v = 0; v < 8; ++v)
        outp[(size_t)(mrow0 + v) * DIM + n] = (_Float16)(accs[j][v] + bn);
    }
  }
}

// ---------------- scores = scale * q @ k^T  (per batch) ---------------------
__global__ void __launch_bounds__(256)
k_scores(const _Float16* __restrict__ qh,
         const _Float16* __restrict__ kh,
         float* __restrict__ w) {
  __shared__ _Float16 ldsA[2][64 * 32];
  __shared__ _Float16 ldsB[2][128 * 32];
  const int tid = threadIdx.x, lane = tid & 31, wvi = tid >> 5;
  const int mw = wvi & 3, nw = wvi >> 2;
  const int b  = blockIdx.z;
  const int Qb = blockIdx.x * 64;
  const int Kb = blockIdx.y * 128;
  const int l15 = lane & 15;
  const int khalf = (lane < 16) ? 0 : 16;
  const float scale = 0.03125f;  // 1/sqrt(1024)
  const uint32_t A_STRIDE = 64 * 32 * 2;
  const uint32_t B_STRIDE = 128 * 32 * 2;
  const uint32_t ldsABase = (uint32_t)(uintptr_t)&ldsA[0][0];
  const uint32_t ldsBBase = (uint32_t)(uintptr_t)&ldsB[0][0];
  const _Float16* ldsAp = &ldsA[0][0];
  const _Float16* ldsBp = &ldsB[0][0];

  v8f acc0 = {}, acc1 = {}, acc2 = {}, acc3 = {};
  const _Float16* qRow0 = qh + (size_t)(b * SEQ + Qb) * DIM;
  const _Float16* kRow0 = kh + (size_t)(b * SEQ + Kb) * DIM;  // k rows = B cols

  auto issueAB = [&](int kb, uint32_t wrA, uint32_t wrB) {
    {
      const int row = tid >> 2, ch = tid & 3;
      const _Float16* srcA = qRow0 + (size_t)row * DIM + kb * KBLK + ch * 8;
      async_b128(ldsABase + wrA + (uint32_t)(row * 64 + ch * 16), srcA);
    }
    {
      const int c = tid >> 1, kk0 = (tid & 1) * 16;
      const _Float16* srcB = kRow0 + (size_t)c * DIM + kb * KBLK + kk0;
      uint32_t offB = ldsBBase + wrB + (uint32_t)(c * 64 + kk0 * 2);
      async_b128(offB, srcB); async_b128(offB + 16, srcB + 8);
    }
  };

  const int NK = DIM / KBLK;
  uint32_t wrA = 0, wrB = 0, rdA = 0, rdB = 0;
  issueAB(0, 0, 0);
  const uint32_t aFragIdx = (uint32_t)((mw * 16 + l15) * 32 + khalf);
  const uint32_t bFragIdx = (uint32_t)(nw * 64 * 32 + khalf);
#pragma clang loop unroll(disable)
  for (int kb = 0; kb < NK; ++kb) {
    wait_async0();
    __syncthreads();
    wrA ^= A_STRIDE; wrB ^= B_STRIDE;
    if (kb + 1 < NK) issueAB(kb + 1, wrA, wrB);
    v16h a = *reinterpret_cast<const v16h*>(ldsAp + rdA + aFragIdx);
    const _Float16* lb = ldsBp + rdB + bFragIdx;
    v16h b0 = *reinterpret_cast<const v16h*>(lb + (0 * 16 + l15) * 32);
    v16h b1 = *reinterpret_cast<const v16h*>(lb + (1 * 16 + l15) * 32);
    v16h b2 = *reinterpret_cast<const v16h*>(lb + (2 * 16 + l15) * 32);
    v16h b3 = *reinterpret_cast<const v16h*>(lb + (3 * 16 + l15) * 32);
    acc0 = wmma_step(a, b0, acc0);
    acc1 = wmma_step(a, b1, acc1);
    acc2 = wmma_step(a, b2, acc2);
    acc3 = wmma_step(a, b3, acc3);
    rdA ^= (A_STRIDE / 2); rdB ^= (B_STRIDE / 2);
  }

  const int q0 = Qb + mw * 16 + ((lane < 16) ? 0 : 8);
  float* wb = w + (size_t)b * SEQ * SEQ;
  v8f accs[4] = {acc0, acc1, acc2, acc3};
#pragma unroll
  for (int j = 0; j < 4; ++j) {
    const int kcol = Kb + nw * 64 + j * 16 + l15;
#pragma unroll
    for (int v = 0; v < 8; ++v)
      wb[(size_t)(q0 + v) * SEQ + kcol] = accs[j][v] * scale;
  }
}

// ----- softmax over QUERY axis (axis=1): per (b, kcol) column, stride S -----
__global__ void __launch_bounds__(256) k_softmax_q(float* __restrict__ w) {
  __shared__ float sm[4][64], ss[4][64];
  const int tid = threadIdx.x;
  const int c   = tid & 63;
  const int r   = tid >> 6;              // 0..3
  const int b   = blockIdx.y;
  const int col = blockIdx.x * 64 + c;
  float* base = w + (size_t)b * SEQ * SEQ + col;

  float m = -INFINITY, s = 0.f;
  for (int q = r; q < SEQ; q += 4) {
    float x  = base[(size_t)q * SEQ];
    float nm = fmaxf(m, x);
    s = s * __expf(m - nm) + __expf(x - nm);
    m = nm;
  }
  sm[r][c] = m; ss[r][c] = s;
  __syncthreads();
  float M = fmaxf(fmaxf(sm[0][c], sm[1][c]), fmaxf(sm[2][c], sm[3][c]));
  float S = 0.f;
#pragma unroll
  for (int i = 0; i < 4; ++i) S += ss[i][c] * __expf(sm[i][c] - M);
  const float inv = 1.f / S;
  for (int q = r; q < SEQ; q += 4) {
    float x = base[(size_t)q * SEQ];
    base[(size_t)q * SEQ] = __expf(x - M) * inv;
  }
}

// ------------- out = attn_weights @ v  (A: f32 in LDS -> f16) ---------------
__global__ void __launch_bounds__(256)
k_av(const float* __restrict__ w,
     const _Float16* __restrict__ vP,
     float* __restrict__ out) {
  __shared__ float    ldsAf[2][64 * 32];  // 2 x 8 KB (f32 weights tile)
  __shared__ _Float16 ldsB[2][128 * 32];  // 2 x 8 KB
  const int tid = threadIdx.x, lane = tid & 31, wvi = tid >> 5;
  const int mw = wvi & 3, nw = wvi >> 2;
  const int b  = blockIdx.z;
  const int Qb = blockIdx.x * 64;
  const int Db = blockIdx.y * 128;
  const int l15 = lane & 15;
  const int kq  = (lane < 16) ? 0 : 16;
  const uint32_t A_STRIDE = 64 * 32 * 4;   // f32 buffer bytes
  const uint32_t B_STRIDE = 128 * 32 * 2;
  const uint32_t ldsABase = (uint32_t)(uintptr_t)&ldsAf[0][0];
  const uint32_t ldsBBase = (uint32_t)(uintptr_t)&ldsB[0][0];
  const float*    ldsAp = &ldsAf[0][0];
  const _Float16* ldsBp = &ldsB[0][0];

  v8f acc0 = {}, acc1 = {}, acc2 = {}, acc3 = {};
  const float*    aRow0 = w  + (size_t)b * SEQ * SEQ + (size_t)Qb * SEQ;
  const _Float16* vB    = vP + (size_t)b * SEQ * DIM;

  auto issueAB = [&](int kb, uint32_t wrA, uint32_t wrB) {
    {  // A tile f32: 64 rows x 32 floats; 32B chunk per thread (2 x b128)
      const int row = tid >> 2, ch = tid & 3;
      const float* srcA = aRow0 + (size_t)row * SEQ + kb * KBLK + ch * 8;
      uint32_t offA = ldsABase + wrA + (uint32_t)(row * 128 + ch * 32);
      async_b128(offA, srcA); async_b128(offA + 16, srcA + 4);
    }
    {
      const int c = tid >> 1, kk0 = (tid & 1) * 16;
      const _Float16* srcB = vB + (size_t)kb * (DIM * 32) + (size_t)(Db + c) * 32 + kk0;
      uint32_t offB = ldsBBase + wrB + (uint32_t)(c * 64 + kk0 * 2);
      async_b128(offB, srcB); async_b128(offB + 16, srcB + 8);
    }
  };

  const int NK = SEQ / KBLK;  // 64
  uint32_t wrA = 0, wrB = 0, rdA = 0, rdB = 0;   // rdA in floats, rdB in halves
  issueAB(0, 0, 0);
  const uint32_t aFragIdx = (uint32_t)((mw * 16 + l15) * 32 + kq);
  const uint32_t bFragIdx = (uint32_t)(nw * 64 * 32 + kq);
#pragma clang loop unroll(disable)
  for (int kb = 0; kb < NK; ++kb) {
    wait_async0();
    __syncthreads();
    wrA ^= A_STRIDE; wrB ^= B_STRIDE;
    if (kb + 1 < NK) issueAB(kb + 1, wrA, wrB);
    const float4* la = reinterpret_cast<const float4*>(ldsAp + rdA + aFragIdx);
    float4 c0 = la[0], c1 = la[1], c2 = la[2], c3 = la[3];
    v16h a;
    a[0]  = (_Float16)c0.x; a[1]  = (_Float16)c0.y; a[2]  = (_Float16)c0.z; a[3]  = (_Float16)c0.w;
    a[4]  = (_Float16)c1.x; a[5]  = (_Float16)c1.y; a[6]  = (_Float16)c1.z; a[7]  = (_Float16)c1.w;
    a[8]  = (_Float16)c2.x; a[9]  = (_Float16)c2.y; a[10] = (_Float16)c2.z; a[11] = (_Float16)c2.w;
    a[12] = (_Float16)c3.x; a[13] = (_Float16)c3.y; a[14] = (_Float16)c3.z; a[15] = (_Float16)c3.w;
    const _Float16* lb = ldsBp + rdB + bFragIdx;
    v16h b0 = *reinterpret_cast<const v16h*>(lb + (0 * 16 + l15) * 32);
    v16h b1 = *reinterpret_cast<const v16h*>(lb + (1 * 16 + l15) * 32);
    v16h b2 = *reinterpret_cast<const v16h*>(lb + (2 * 16 + l15) * 32);
    v16h b3 = *reinterpret_cast<const v16h*>(lb + (3 * 16 + l15) * 32);
    acc0 = wmma_step(a, b0, acc0);
    acc1 = wmma_step(a, b1, acc1);
    acc2 = wmma_step(a, b2, acc2);
    acc3 = wmma_step(a, b3, acc3);
    rdA ^= (A_STRIDE / 4); rdB ^= (B_STRIDE / 2);   // element-unit toggles
  }

  const int q0 = Qb + mw * 16 + ((lane < 16) ? 0 : 8);
  v8f accs[4] = {acc0, acc1, acc2, acc3};
#pragma unroll
  for (int j = 0; j < 4; ++j) {
    const int n = Db + nw * 64 + j * 16 + l15;
#pragma unroll
    for (int v = 0; v < 8; ++v)
      out[(size_t)(b * SEQ + q0 + v) * DIM + n] = accs[j][v];
  }
}

extern "C" void kernel_launch(void* const* d_in, const int* in_sizes, int n_in,
                              void* d_out, int out_size, void* d_ws, size_t ws_size,
                              hipStream_t stream) {
  const float* x  = (const float*)d_in[0];
  const float* Wq = (const float*)d_in[1];
  const float* bq = (const float*)d_in[2];
  const float* Wk = (const float*)d_in[3];
  const float* bk = (const float*)d_in[4];
  const float* Wv = (const float*)d_in[5];
  const float* bv = (const float*)d_in[6];

  float* out = (float*)d_out;                 // attn_output: BS*DIM f32
  float* wgt = out + (size_t)BS * DIM;        // attn_weights: B*S*S f32

  _Float16* xh  = (_Float16*)d_ws;
  _Float16* wqP = xh  + (size_t)BS * DIM;
  _Float16* wkP = wqP + (size_t)DIM * DIM;
  _Float16* wvP = wkP + (size_t)DIM * DIM;
  _Float16* qh  = wvP + (size_t)DIM * DIM;
  _Float16* kh  = qh  + (size_t)BS * DIM;
  _Float16* vP  = kh  + (size_t)BS * DIM;

  k_cvt_x<<<(BS * DIM) / (256 * 8), 256, 0, stream>>>(x, xh);
  k_cvt_w_packed<<<(DIM * (DIM / 32)) / 256, 256, 0, stream>>>(Wq, wqP);
  k_cvt_w_packed<<<(DIM * (DIM / 32)) / 256, 256, 0, stream>>>(Wk, wkP);
  k_cvt_w_packed<<<(DIM * (DIM / 32)) / 256, 256, 0, stream>>>(Wv, wvP);

  dim3 gQKV(BS / 64, DIM / 128);
  k_qkv_gemm<false><<<gQKV, 256, 0, stream>>>(xh, wqP, bq, qh);
  k_qkv_gemm<false><<<gQKV, 256, 0, stream>>>(xh, wkP, bk, kh);
  k_qkv_gemm<true ><<<gQKV, 256, 0, stream>>>(xh, wvP, bv, vP);

  dim3 gS(SEQ / 64, SEQ / 128, BATCH);
  k_scores<<<gS, 256, 0, stream>>>(qh, kh, wgt);

  dim3 gSm(SEQ / 64, BATCH);
  k_softmax_q<<<gSm, 256, 0, stream>>>(wgt);

  dim3 gAV(SEQ / 64, DIM / 128, BATCH);
  k_av<<<gAV, 256, 0, stream>>>(wgt, vP, out);
}
